// GraphEncoder_4612794876702
// MI455X (gfx1250) — matrix operand order repaired and checked by reading, not compiled, not run on hardware
//
#include <hip/hip_runtime.h>

// ---------------------------------------------------------------------------
// GraphEncoder (GCN-VAE) for gfx1250:  bf16 WMMA pipeline
//   d = rsqrt(rowsum(A)+1+1e-8)
//   h  = d[n]*(A @ (d*x))[n] + d[n]^2 * x[n]          (identity folded)
//   h1 = relu(h @ W1^T + b1)
//   mu = h1 @ Wmu^T + bmu ; logvar = h1 @ Wlv^T + blv
// Per-wave tile 32x64 (2x4 WMMA accum), block tile 64x256 (8 waves).
// ---------------------------------------------------------------------------

typedef __bf16 bf16;
typedef __attribute__((ext_vector_type(8)))  bf16  v8bf;
typedef __attribute__((ext_vector_type(16))) bf16  v16bf;
typedef __attribute__((ext_vector_type(8)))  float v8f;

#define BATCH 64
#define NNODE 1024
#define DIN   512
#define HID   1024
#define LAT   256

// ---- WMMA fragment loaders (wave32, ISA 7.12.2 layouts) --------------------
// A (16x32, 16-bit): lanes 0-15 row M=lane hold K={0..7,16..23};
//                    lanes 16-31 row M=lane-16 hold K={8..15,24..31}.
__device__ __forceinline__ v16bf frag_a(const bf16* __restrict__ base, int ld) {
    int lane = threadIdx.x & 31;
    const bf16* p = base + (size_t)(lane & 15) * ld + ((lane >> 4) << 3);
    v8bf lo = *(const v8bf*)p;
    v8bf hi = *(const v8bf*)(p + 16);
    v16bf v;
#pragma unroll
    for (int i = 0; i < 8; ++i) { v[i] = lo[i]; v[i + 8] = hi[i]; }
    return v;
}

// Same as frag_a but converts f32 source (adjacency) to bf16 on the fly.
__device__ __forceinline__ v16bf frag_a_f32(const float* __restrict__ base, int ld) {
    int lane = threadIdx.x & 31;
    const float* p = base + (size_t)(lane & 15) * ld + ((lane >> 4) << 3);
    v16bf v;
#pragma unroll
    for (int i = 0; i < 8; ++i) { v[i] = (bf16)p[i]; v[i + 8] = (bf16)p[i + 16]; }
    return v;
}

// B (32x16, 16-bit): lanes 0-15 col N=lane hold K={0..7,16..23};
//                    lanes 16-31 col N=lane-16 hold K={8..15,24..31}.
// `base` points to row-major B^T (i.e. weight[outcol][k]), ld = K-stride.
__device__ __forceinline__ v16bf frag_b(const bf16* __restrict__ base, int ld) {
    int lane = threadIdx.x & 31;
    const bf16* p = base + (size_t)(lane & 15) * ld + ((lane >> 4) << 3);
    v8bf lo = *(const v8bf*)p;
    v8bf hi = *(const v8bf*)(p + 16);
    v16bf v;
#pragma unroll
    for (int i = 0; i < 8; ++i) { v[i] = lo[i]; v[i + 8] = hi[i]; }
    return v;
}

// ---- degree / normalization scale: one wave per adjacency row --------------
__global__ void k_deg(const float* __restrict__ a, float* __restrict__ dbuf) {
    int lane = threadIdx.x & 31;
    int wave = threadIdx.x >> 5;
    int row  = blockIdx.x * 8 + wave;            // 0 .. BATCH*NNODE-1
    const float* p = a + (size_t)row * NNODE;
    float s = 0.f;
#pragma unroll
    for (int i = 0; i < NNODE / 32; ++i) s += p[lane + i * 32];
#pragma unroll
    for (int o = 16; o > 0; o >>= 1) s += __shfl_xor(s, o, 32);
    if (lane == 0) dbuf[row] = rsqrtf(s + 1.0f + 1e-8f);   // deg includes +I
}

// ---- f32 -> bf16 weight conversion -----------------------------------------
__global__ void k_f32_to_bf16(const float* __restrict__ in, bf16* __restrict__ out, int n) {
    int i = blockIdx.x * blockDim.x + threadIdx.x;
    if (i < n) out[i] = (bf16)in[i];
}

// ---- xs^T[b][c][m] = bf16(d[b,m] * x[b,m,c])  (N-major for B-frag loads) ---
__global__ void k_make_xst(const float* __restrict__ x, const float* __restrict__ dbuf,
                           bf16* __restrict__ xst) {
    int idx = blockIdx.x * 256 + threadIdx.x;    // b<<19 | c<<10 | m
    int m = idx & (NNODE - 1);
    int c = (idx >> 10) & (DIN - 1);
    int b = idx >> 19;
    float dv = dbuf[b * NNODE + m];
    float xv = x[((size_t)(b * NNODE + m)) * DIN + c];
    xst[idx] = (bf16)(dv * xv);
}

// ---- GEMM1: h' = A(f32->bf16) @ xs^T, epilogue h = d*h' + d^2*x ------------
__global__ void k_gemm_adj(const float* __restrict__ a, const float* __restrict__ x,
                           const float* __restrict__ dbuf, const bf16* __restrict__ xst,
                           bf16* __restrict__ hbuf) {
    const int b = blockIdx.z;
    const float* A  = a   + (size_t)b * NNODE * NNODE;
    const bf16*  Bx = xst + (size_t)b * DIN * NNODE;
    int lane = threadIdx.x & 31;
    int wave = threadIdx.x >> 5;
    int m0 = blockIdx.y * 64  + (wave >> 2) * 32;   // node rows
    int n0 = blockIdx.x * 256 + (wave & 3) * 64;    // feature cols

    v8f zero = {0.f, 0.f, 0.f, 0.f, 0.f, 0.f, 0.f, 0.f};
    v8f acc[2][4];
#pragma unroll
    for (int i = 0; i < 2; ++i)
#pragma unroll
        for (int j = 0; j < 4; ++j) acc[i][j] = zero;

    for (int k0 = 0; k0 < NNODE; k0 += 32) {
        v16bf af[2], bfr[4];
#pragma unroll
        for (int i = 0; i < 2; ++i)
            af[i] = frag_a_f32(A + (size_t)(m0 + 16 * i) * NNODE + k0, NNODE);
#pragma unroll
        for (int j = 0; j < 4; ++j)
            bfr[j] = frag_b(Bx + (size_t)(n0 + 16 * j) * NNODE + k0, NNODE);
#pragma unroll
        for (int i = 0; i < 2; ++i)
#pragma unroll
            for (int j = 0; j < 4; ++j)
                acc[i][j] = __builtin_amdgcn_wmma_f32_16x16x32_bf16(
                    false, af[i], false, bfr[j], (short)0, acc[i][j], false, false);
    }

    int cl = lane & 15, rh = (lane >> 4) << 3;
#pragma unroll
    for (int i = 0; i < 2; ++i)
#pragma unroll
        for (int j = 0; j < 4; ++j) {
            int col = n0 + j * 16 + cl;
            int rb  = m0 + i * 16 + rh;
#pragma unroll
            for (int r = 0; r < 8; ++r) {
                int row  = rb + r;
                float dv = dbuf[b * NNODE + row];
                float xv = x[((size_t)(b * NNODE + row)) * DIN + col];
                float v  = dv * acc[i][j][r] + dv * dv * xv;   // +I term
                hbuf[((size_t)(b * NNODE + row)) * DIN + col] = (bf16)v;
            }
        }
}

// ---- generic bf16 GEMM: out = act(A @ Bw^T + bias) -------------------------
template <bool RELU, bool OUTBF>
__global__ void k_gemm(const bf16* __restrict__ A, int lda,
                       const bf16* __restrict__ Bw, int ldb,
                       const float* __restrict__ bias, int K,
                       bf16* __restrict__ outb, float* __restrict__ outf, int ldo) {
    const int b = blockIdx.z;
    A += (size_t)b * NNODE * lda;
    if (OUTBF) outb += (size_t)b * NNODE * ldo;
    else       outf += (size_t)b * NNODE * ldo;

    int lane = threadIdx.x & 31;
    int wave = threadIdx.x >> 5;
    int m0 = blockIdx.y * 64  + (wave >> 2) * 32;
    int n0 = blockIdx.x * 256 + (wave & 3) * 64;

    v8f zero = {0.f, 0.f, 0.f, 0.f, 0.f, 0.f, 0.f, 0.f};
    v8f acc[2][4];
#pragma unroll
    for (int i = 0; i < 2; ++i)
#pragma unroll
        for (int j = 0; j < 4; ++j) acc[i][j] = zero;

    for (int k0 = 0; k0 < K; k0 += 32) {
        v16bf af[2], bfr[4];
#pragma unroll
        for (int i = 0; i < 2; ++i)
            af[i] = frag_a(A + (size_t)(m0 + 16 * i) * lda + k0, lda);
#pragma unroll
        for (int j = 0; j < 4; ++j)
            bfr[j] = frag_b(Bw + (size_t)(n0 + 16 * j) * ldb + k0, ldb);
#pragma unroll
        for (int i = 0; i < 2; ++i)
#pragma unroll
            for (int j = 0; j < 4; ++j)
                acc[i][j] = __builtin_amdgcn_wmma_f32_16x16x32_bf16(
                    false, af[i], false, bfr[j], (short)0, acc[i][j], false, false);
    }

    int cl = lane & 15, rh = (lane >> 4) << 3;
#pragma unroll
    for (int i = 0; i < 2; ++i)
#pragma unroll
        for (int j = 0; j < 4; ++j) {
            int col = n0 + j * 16 + cl;
            int rb  = m0 + i * 16 + rh;
            float bv = bias[col];
#pragma unroll
            for (int r = 0; r < 8; ++r) {
                float v = acc[i][j][r] + bv;
                if (RELU) v = fmaxf(v, 0.f);
                if (OUTBF) outb[(size_t)(rb + r) * ldo + col] = (bf16)v;
                else       outf[(size_t)(rb + r) * ldo + col] = v;
            }
        }
}

// ---------------------------------------------------------------------------
extern "C" void kernel_launch(void* const* d_in, const int* in_sizes, int n_in,
                              void* d_out, int out_size, void* d_ws, size_t ws_size,
                              hipStream_t stream) {
    const float* x   = (const float*)d_in[0];
    const float* a   = (const float*)d_in[1];
    const float* W1  = (const float*)d_in[2];
    const float* b1  = (const float*)d_in[3];
    const float* Wmu = (const float*)d_in[4];
    const float* bmu = (const float*)d_in[5];
    const float* Wlv = (const float*)d_in[6];
    const float* blv = (const float*)d_in[7];
    float* out = (float*)d_out;
    (void)in_sizes; (void)n_in; (void)out_size; (void)ws_size;

    // workspace layout (~258.3 MB, all offsets 256B-aligned)
    char* ws = (char*)d_ws;
    size_t off = 0;
    float* dbuf = (float*)(ws + off); off += (size_t)BATCH * NNODE * sizeof(float);
    bf16* xst   = (bf16*)(ws + off);  off += (size_t)BATCH * DIN * NNODE * sizeof(bf16);
    bf16* W1b   = (bf16*)(ws + off);  off += (size_t)HID * DIN * sizeof(bf16);
    bf16* Wmub  = (bf16*)(ws + off);  off += (size_t)LAT * HID * sizeof(bf16);
    bf16* Wlvb  = (bf16*)(ws + off);  off += (size_t)LAT * HID * sizeof(bf16);
    bf16* hbuf  = (bf16*)(ws + off);  off += (size_t)BATCH * NNODE * DIN * sizeof(bf16);
    bf16* h1buf = (bf16*)(ws + off);  off += (size_t)BATCH * NNODE * HID * sizeof(bf16);

    k_deg<<<BATCH * NNODE / 8, 256, 0, stream>>>(a, dbuf);
    k_f32_to_bf16<<<(HID * DIN) / 256, 256, 0, stream>>>(W1,  W1b,  HID * DIN);
    k_f32_to_bf16<<<(LAT * HID) / 256, 256, 0, stream>>>(Wmu, Wmub, LAT * HID);
    k_f32_to_bf16<<<(LAT * HID) / 256, 256, 0, stream>>>(Wlv, Wlvb, LAT * HID);
    k_make_xst<<<(BATCH * DIN * NNODE) / 256, 256, 0, stream>>>(x, dbuf, xst);

    // h = d*(A @ xs) + d^2*x          [B,1024,512] bf16
    k_gemm_adj<<<dim3(DIN / 256, NNODE / 64, BATCH), 256, 0, stream>>>(a, x, dbuf, xst, hbuf);
    // h1 = relu(h @ W1^T + b1)        [B,1024,1024] bf16
    k_gemm<true, true><<<dim3(HID / 256, NNODE / 64, BATCH), 256, 0, stream>>>(
        hbuf, DIN, W1b, DIN, b1, DIN, h1buf, nullptr, HID);
    // mu / logvar                     [B,1024,256] f32 -> d_out
    k_gemm<false, false><<<dim3(LAT / 256, NNODE / 64, BATCH), 256, 0, stream>>>(
        h1buf, HID, Wmub, HID, bmu, HID, nullptr, out, LAT);
    k_gemm<false, false><<<dim3(LAT / 256, NNODE / 64, BATCH), 256, 0, stream>>>(
        h1buf, HID, Wlvb, HID, blv, HID, nullptr, out + (size_t)BATCH * NNODE * LAT, LAT);
}
